// LSTMClassifier_65816078844132
// MI455X (gfx1250) — matrix operand order repaired and checked by reading, not compiled, not run on hardware
//
#include <hip/hip_runtime.h>

typedef __attribute__((ext_vector_type(16))) __bf16        v16bf;
typedef __attribute__((ext_vector_type(8)))  float         v8f;
typedef __attribute__((ext_vector_type(4)))  unsigned int  u32x4;

union FragBF {
  v16bf v;
  u32x4 q[2];
};

__device__ __forceinline__ unsigned short f2bf(float f) {
  unsigned u = __builtin_bit_cast(unsigned, f);
  u += 0x7fffu + ((u >> 16) & 1u);     // round-to-nearest-even
  return (unsigned short)(u >> 16);
}
__device__ __forceinline__ float sigf(float x) { return 1.0f / (1.0f + __expf(-x)); }

static constexpr int Bn = 64, Tn = 512, En = 256, Hn = 256, G4H = 1024, Cn = 50;

// ---------- embedding gather: x0[bt,e] = bf16(emb[text[bt], e]) ----------
__global__ __launch_bounds__(256) void gather_embed(const int* __restrict__ text,
                                                    const float* __restrict__ emb,
                                                    unsigned short* __restrict__ x0) {
  int idx = blockIdx.x * 256 + threadIdx.x;      // over B*T*E
  int e   = idx & (En - 1);
  int bt  = idx >> 8;
  int tok = text[bt];
  x0[idx] = f2bf(emb[(size_t)tok * En + e]);
}

// ---------- pre-swizzle W[N=1024, K] (row-major f32, gates = x * W^T) into
// B-fragment order: P[((tn*TK+tk)*32 + lane)*16 + j] = bf16(W[tn*16+(lane&15)][tk*32+8*(lane>>4)+j+(j&8)])
__global__ __launch_bounds__(256) void pack_w(const float* __restrict__ W,
                                              unsigned short* __restrict__ P, int K) {
  int idx = blockIdx.x * 256 + threadIdx.x;
  int TK = K >> 5;
  if (idx >= 64 * TK * 32) return;
  int lane = idx & 31;
  int tile = idx >> 5;                 // == tn*TK + tk
  int tk = tile % TK;
  int tn = tile / TK;
  int n  = tn * 16 + (lane & 15);
  int kb = tk * 32 + ((lane >> 4) << 3);
  const float* wr = W + (size_t)n * K + kb;
  unsigned short* o = P + (size_t)idx * 16;
#pragma unroll
  for (int j = 0; j < 16; ++j) o[j] = f2bf(wr[j + (j & 8)]);
}

__global__ __launch_bounds__(256) void bias_comb_k(const float* __restrict__ a,
                                                   const float* __restrict__ b,
                                                   float* __restrict__ o) {
  int i = blockIdx.x * 256 + threadIdx.x;
  if (i < G4H) o[i] = a[i] + b[i];
}

// ---------- big input GEMM: Cout (time-major [T][B][N]) = bf16(A[MxK]) * packedB + bias ----------
// A rows are (b*T + t); the store transposes to (t*B + b) so the scan's per-step
// 256KB block is contiguous. Stores are non-temporal (written once, read once).
__global__ __launch_bounds__(256) void gemm_bf16_bias(const unsigned short* __restrict__ A,
                                                      const unsigned short* __restrict__ Bp,
                                                      const float* __restrict__ bias,
                                                      float* __restrict__ C,
                                                      int M, int N, int K) {
  const int lane = threadIdx.x & 31;
  const int wave = threadIdx.x >> 5;
  const int lhi = lane >> 4, llo = lane & 15;
  const int ntn = N >> 4;
  const int tile = blockIdx.x * 8 + wave;
  const int mt = tile / ntn, nt = tile - mt * ntn;
  const int m0 = mt << 4, n0 = nt << 4;
  const int TK = K >> 5;
  v8f acc = {0.f, 0.f, 0.f, 0.f, 0.f, 0.f, 0.f, 0.f};
  const unsigned short* arow = A + (size_t)(m0 + llo) * K + (lhi << 3);
  for (int tk = 0; tk < TK; ++tk) {
    FragBF a, b;
    const u32x4* ap = (const u32x4*)(arow + (tk << 5));
    a.q[0] = ap[0];                   // K halves +0..7   (VGPR 0-3)
    a.q[1] = ap[2];                   // K halves +16..23 (VGPR 4-7)
    const u32x4* bp = (const u32x4*)(Bp + (((size_t)nt * TK + tk) * 32 + lane) * 16);
    b.q[0] = bp[0];
    b.q[1] = bp[1];
    acc = __builtin_amdgcn_wmma_f32_16x16x32_bf16(false, a.v, false, b.v,
                                                  (short)0, acc, false, false);
  }
  const float bc = bias[n0 + llo];
#pragma unroll
  for (int r = 0; r < 8; ++r) {
    const int row = m0 + r + (lhi << 3);      // = b*Tn + t
    const int b_  = row >> 9;                 // Tn == 512
    const int t_  = row & (Tn - 1);
    __builtin_nontemporal_store(acc[r] + bc,
        C + ((size_t)t_ * Bn + b_) * N + n0 + llo);
  }
}

// ---------- sequential masked LSTM scan, one workgroup per direction ----------
// xg is time-major [T][B][4H]; next step's block is prefetched during compute.
__global__ __launch_bounds__(1024) void lstm_scan(const float* __restrict__ xg_f,
                                                  const float* __restrict__ xg_b,
                                                  const unsigned short* __restrict__ whhp_f,
                                                  const unsigned short* __restrict__ whhp_b,
                                                  const int* __restrict__ lens,
                                                  unsigned short* __restrict__ y,
                                                  float* __restrict__ hfin_f,
                                                  float* __restrict__ hfin_b,
                                                  int store_y) {
  __shared__ __attribute__((aligned(16))) unsigned short hlds[2][Bn * Hn];  // bf16 h, double-buffered
  const int dir = blockIdx.x;
  const float* xg = dir ? xg_b : xg_f;
  const unsigned short* whhp = dir ? whhp_b : whhp_f;
  float* hfin = dir ? hfin_b : hfin_f;
  const int ycol = dir ? Hn : 0;
  const int lane = threadIdx.x & 31, wave = threadIdx.x >> 5;
  const int lhi = lane >> 4, llo = lane & 15;

  int mb[2], ns[2], lenr[2][8];
  float cst[2][8], hst[2][8];
#pragma unroll
  for (int j = 0; j < 2; ++j) {
    const int gi = wave + 32 * j;          // 64 (m-block, hidden-slice) groups
    mb[j] = gi >> 4;
    ns[j] = gi & 15;
#pragma unroll
    for (int r = 0; r < 8; ++r) {
      cst[j][r] = 0.f;
      hst[j][r] = 0.f;
      lenr[j][r] = lens[mb[j] * 16 + r + (lhi << 3)];
    }
  }
  for (int i = threadIdx.x; i < Bn * Hn; i += 1024) hlds[0][i] = 0;
  __syncthreads();

  for (int t = 0; t < Tn; ++t) {
    const int tt = dir ? (Tn - 1 - t) : t;
    const unsigned short* hr = hlds[t & 1];
    unsigned short* hw = hlds[(t + 1) & 1];

    // Prefetch next step's contiguous 256KB gate block while this step computes.
    if (t + 1 < Tn) {
      const float* nb = xg + (size_t)(dir ? tt - 1 : tt + 1) * (Bn * G4H)
                        + threadIdx.x * 64;          // 1024 thr * 64 floats = full block
      __builtin_prefetch(nb, 0, 1);
      __builtin_prefetch(nb + 32, 0, 1);             // second 128B line
    }

#pragma unroll
    for (int j = 0; j < 2; ++j) {
      const int b0 = mb[j] << 4;
      const int nsj = ns[j];
      v8f acc[4];
      // seed accumulators with xg (already has bih+bhh folded in); NT: read-once stream
#pragma unroll
      for (int g = 0; g < 4; ++g)
#pragma unroll
        for (int r = 0; r < 8; ++r)
          acc[g][r] = __builtin_nontemporal_load(
              xg + (size_t)tt * (Bn * G4H) + (b0 + r + (lhi << 3)) * G4H
                 + g * Hn + nsj * 16 + llo);
      // gates += h * Whh^T : 8 k-steps, A fragment reused across the 4 gates
#pragma unroll
      for (int tk = 0; tk < 8; ++tk) {
        FragBF a;
        const u32x4* ap = (const u32x4*)(hr + (b0 + llo) * Hn + (tk << 5) + (lhi << 3));
        a.q[0] = ap[0];
        a.q[1] = ap[2];
#pragma unroll
        for (int g = 0; g < 4; ++g) {
          const int tn = g * 16 + nsj;
          FragBF b;
          const u32x4* bp = (const u32x4*)(whhp + ((size_t)(tn * 8 + tk) * 32 + lane) * 16);
          b.q[0] = bp[0];
          b.q[1] = bp[1];
          acc[g] = __builtin_amdgcn_wmma_f32_16x16x32_bf16(false, a.v, false, b.v,
                                                           (short)0, acc[g], false, false);
        }
      }
      // cell update in registers; masked state freeze
#pragma unroll
      for (int r = 0; r < 8; ++r) {
        const int brow = b0 + r + (lhi << 3);
        const int n = nsj * 16 + llo;
        const bool msk = tt < lenr[j][r];
        const float ig = sigf(acc[0][r]);
        const float fg = sigf(acc[1][r]);
        const float gg = tanhf(acc[2][r]);
        const float og = sigf(acc[3][r]);
        const float cn = fg * cst[j][r] + ig * gg;
        const float hn = og * tanhf(cn);
        if (msk) { cst[j][r] = cn; hst[j][r] = hn; }
        hw[brow * Hn + n] = f2bf(hst[j][r]);
        if (store_y)
          y[((size_t)brow * Tn + tt) * (2 * Hn) + ycol + n] = msk ? f2bf(hn) : (unsigned short)0;
      }
    }
    __syncthreads();   // next-buffer writes complete; prev-buffer reads complete
  }
#pragma unroll
  for (int j = 0; j < 2; ++j)
#pragma unroll
    for (int r = 0; r < 8; ++r)
      hfin[(size_t)(mb[j] * 16 + r + (lhi << 3)) * Hn + ns[j] * 16 + llo] = hst[j][r];
}

// ---------- tiny FC head: out[b,c] = concat(h0b,h1b)[b,:] . fcW[c,:] + fcb[c] ----------
__global__ __launch_bounds__(64) void fc_head(const float* __restrict__ h0b,
                                              const float* __restrict__ h1b,
                                              const float* __restrict__ W,
                                              const float* __restrict__ bias,
                                              float* __restrict__ out) {
  const int c = blockIdx.x;   // 0..49
  const int b = threadIdx.x;  // 0..63
  float s = bias[c];
  const float* wr = W + (size_t)c * (2 * Hn);
  for (int k = 0; k < Hn; ++k) s += h0b[b * Hn + k] * wr[k];
  for (int k = 0; k < Hn; ++k) s += h1b[b * Hn + k] * wr[Hn + k];
  out[b * Cn + c] = s;
}

extern "C" void kernel_launch(void* const* d_in, const int* in_sizes, int n_in,
                              void* d_out, int out_size, void* d_ws, size_t ws_size,
                              hipStream_t stream) {
  (void)in_sizes; (void)n_in; (void)out_size; (void)ws_size;
  const int*   text = (const int*)d_in[0];
  const int*   lens = (const int*)d_in[1];
  const float* emb  = (const float*)d_in[2];
  // order: l0f, l0b, l1f, l1b
  const float* Wih[4] = {(const float*)d_in[3],  (const float*)d_in[7],
                         (const float*)d_in[11], (const float*)d_in[15]};
  const float* Whh[4] = {(const float*)d_in[4],  (const float*)d_in[8],
                         (const float*)d_in[12], (const float*)d_in[16]};
  const float* bih[4] = {(const float*)d_in[5],  (const float*)d_in[9],
                         (const float*)d_in[13], (const float*)d_in[17]};
  const float* bhh[4] = {(const float*)d_in[6],  (const float*)d_in[10],
                         (const float*)d_in[14], (const float*)d_in[18]};
  const float* fcW = (const float*)d_in[19];
  const float* fcb = (const float*)d_in[20];

  char* base = (char*)d_ws;
  size_t off = 0;
  auto take = [&](size_t bytes) -> char* {
    char* p = base + off;
    off = (off + bytes + 255) & ~(size_t)255;
    return p;
  };
  unsigned short* x0  = (unsigned short*)take((size_t)Bn * Tn * En * 2);
  unsigned short* y0  = (unsigned short*)take((size_t)Bn * Tn * 2 * Hn * 2);
  float* xgf = (float*)take((size_t)Bn * Tn * G4H * 4);
  float* xgb = (float*)take((size_t)Bn * Tn * G4H * 4);
  unsigned short* wihp[4];
  wihp[0] = (unsigned short*)take((size_t)64 * 8  * 32 * 16 * 2);   // K=256
  wihp[1] = (unsigned short*)take((size_t)64 * 8  * 32 * 16 * 2);
  wihp[2] = (unsigned short*)take((size_t)64 * 16 * 32 * 16 * 2);   // K=512
  wihp[3] = (unsigned short*)take((size_t)64 * 16 * 32 * 16 * 2);
  unsigned short* whhp[4];
  float* bc[4];
  float* hfin[4];
  for (int i = 0; i < 4; ++i) whhp[i] = (unsigned short*)take((size_t)64 * 8 * 32 * 16 * 2);
  for (int i = 0; i < 4; ++i) bc[i]   = (float*)take((size_t)G4H * 4);
  for (int i = 0; i < 4; ++i) hfin[i] = (float*)take((size_t)Bn * Hn * 4);

  // 1) gather + weight pre-swizzle (all independent)
  gather_embed<<<(Bn * Tn * En) / 256, 256, 0, stream>>>(text, emb, x0);
  pack_w<<<64,  256, 0, stream>>>(Wih[0], wihp[0], 256);
  pack_w<<<64,  256, 0, stream>>>(Wih[1], wihp[1], 256);
  pack_w<<<128, 256, 0, stream>>>(Wih[2], wihp[2], 512);
  pack_w<<<128, 256, 0, stream>>>(Wih[3], wihp[3], 512);
  for (int i = 0; i < 4; ++i) {
    pack_w<<<64, 256, 0, stream>>>(Whh[i], whhp[i], 256);
    bias_comb_k<<<4, 256, 0, stream>>>(bih[i], bhh[i], bc[i]);
  }

  const int M = Bn * Tn;                       // 32768
  const int tiles = (M / 16) * (G4H / 16);     // 131072, /8 waves per block
  // 2) layer 0: input GEMMs then scan (f,b in parallel workgroups)
  gemm_bf16_bias<<<tiles / 8, 256, 0, stream>>>(x0, wihp[0], bc[0], xgf, M, G4H, En);
  gemm_bf16_bias<<<tiles / 8, 256, 0, stream>>>(x0, wihp[1], bc[1], xgb, M, G4H, En);
  lstm_scan<<<2, 1024, 0, stream>>>(xgf, xgb, whhp[0], whhp[1], lens, y0, hfin[0], hfin[1], 1);
  // 3) layer 1 (input = concat(yf,yb), K=512); y not needed, only final h
  gemm_bf16_bias<<<tiles / 8, 256, 0, stream>>>(y0, wihp[2], bc[2], xgf, M, G4H, 2 * Hn);
  gemm_bf16_bias<<<tiles / 8, 256, 0, stream>>>(y0, wihp[3], bc[3], xgb, M, G4H, 2 * Hn);
  lstm_scan<<<2, 1024, 0, stream>>>(xgf, xgb, whhp[2], whhp[3], lens, y0, hfin[2], hfin[3], 0);
  // 4) head: hidden = concat(h_l0_backward, h_l1_backward)
  fc_head<<<Cn, Bn, 0, stream>>>(hfin[1], hfin[3], fcW, fcb, (float*)d_out);
}
